// LSTMTaggerCharLevel_59021440581691
// MI455X (gfx1250) — matrix operand order repaired and checked by reading, not compiled, not run on hardware
//
#include <hip/hip_runtime.h>
#include <hip/hip_bf16.h>

// ---------------- problem dimensions ----------------
#define S_   4096
#define L_   16
#define DW_  128
#define DC_  64
#define H_   256
#define T_   50
#define CV_  53
#define WXLD 192      // DW + DC
#define G4H  1024     // 4*H
#define G4C  256      // 4*DC

// ---------------- workspace layout (bytes) ----------------
#define WS_WX  0                          // ushort  [S_*WXLD]   bf16 word-LSTM input
#define WS_XG  (WS_WX + S_*WXLD*2)        // float   [S_*G4H]    precomputed x-gates
#define WS_HS  (WS_XG + (size_t)S_*G4H*4) // float   [S_*H_]     word-LSTM hidden states
#define WS_CB  (WS_HS + (size_t)S_*H_*4)  // float   [G4C]       char bias (bih+bhh)
#define WS_WB  (WS_CB + G4C*4)            // float   [G4H]       word bias (bih+bhh)

typedef __attribute__((ext_vector_type(16))) __bf16 v16bf;
typedef __attribute__((ext_vector_type(8)))  float  v8f;
typedef __attribute__((ext_vector_type(4)))  unsigned int u32x4;
typedef __attribute__((ext_vector_type(8)))  int  i32x8;
typedef __attribute__((ext_vector_type(4)))  int  i32x4;

union FragU {
  unsigned short s[16];
  uint4          q[2];
  v16bf          v;
};

__device__ __forceinline__ unsigned short f2bf(float f) {
  return __builtin_bit_cast(unsigned short, (__bf16)f);
}
__device__ __forceinline__ float sigm(float x) { return 1.0f / (1.0f + __expf(-x)); }

// Branch-free tanh: native v_tanh_f32 if available, else exp-based identity
// (the libm tanhf expansion showed s_and_saveexec divergence on the recurrence
//  critical path).
__device__ __forceinline__ float fast_tanh(float x) {
#if defined(__has_builtin) && __has_builtin(__builtin_amdgcn_tanhf)
  return __builtin_amdgcn_tanhf(x);
#else
  return 1.0f - 2.0f / (__expf(2.0f * x) + 1.0f);
#endif
}

// B fragment (32x16 KxN chunk) for G = A * W^T: B[k,n] = W[n][k].
// Lane n (0-15) holds column n with K = kc*32 + 0..7, 16..23; lanes 16-31 hold K = kc*32 + 8..15, 24..31.
__device__ __forceinline__ FragU load_bfrag_f32(const float* __restrict__ W, int ld,
                                                int col, int kc, int lane) {
  FragU b;
  const int kb = kc * 32 + ((lane & 16) ? 8 : 0);
  const float* p = W + (size_t)col * ld + kb;
#pragma unroll
  for (int e = 0; e < 16; ++e) b.s[e] = f2bf(p[e < 8 ? e : e + 8]);
  return b;
}

// A fragment (16x32 MxK chunk) from row-major bf16 storage.
// Lane<16: row=lane, K = kb+0..7 and kb+16..23 (kb = kc*32); lane>=16: row=lane-16, kb += 8.
__device__ __forceinline__ FragU load_afrag_u16(const unsigned short* __restrict__ base,
                                                int ld, int row, int kc, int lane) {
  FragU a;
  const int kb = kc * 32 + ((lane & 16) ? 8 : 0);
  const unsigned short* p = base + (size_t)row * ld + kb;
  a.q[0] = *(const uint4*)(p);
  a.q[1] = *(const uint4*)(p + 16);
  return a;
}

#define WMMA_BF16(A, B, C) \
  __builtin_amdgcn_wmma_f32_16x16x32_bf16(false, (A), false, (B), (short)0, (C), false, false)

// ================= kernel 1: embeddings gather + bias folding =================
__global__ __launch_bounds__(256) void prep_kernel(
    const int* __restrict__ wids, const float* __restrict__ wemb,
    const float* __restrict__ cbih, const float* __restrict__ cbhh,
    const float* __restrict__ wbih, const float* __restrict__ wbhh,
    unsigned short* __restrict__ wx, float* __restrict__ cb, float* __restrict__ wb) {
  const int gid = blockIdx.x * blockDim.x + threadIdx.x;
  const int stride = gridDim.x * blockDim.x;
  for (int i = gid; i < S_ * DW_; i += stride) {
    const int s = i >> 7, d = i & 127;
    wx[(size_t)s * WXLD + d] = f2bf(wemb[(size_t)wids[s] * DW_ + d]);
  }
  if (gid < G4C) cb[gid] = cbih[gid] + cbhh[gid];
  if (gid < G4H) wb[gid] = wbih[gid] + wbhh[gid];
}

// ================= kernel 2: char-level LSTM (WMMA, 16 words / block) =================
__global__ __launch_bounds__(256) void char_lstm_kernel(
    const int* __restrict__ char_ids, const int* __restrict__ word_lengths,
    const float* __restrict__ char_emb,
    const float* __restrict__ cWih, const float* __restrict__ cWhh,
    const float* __restrict__ cb, unsigned short* __restrict__ wx) {
  __shared__ __align__(16) unsigned short ce_lds[CV_ * DC_];
  __shared__ int   cid_lds[16][L_];
  __shared__ int   wlen_lds[16];
  __shared__ float h_lds[16][DC_];
  __shared__ float c_lds[16][DC_];
  __shared__ __align__(16) unsigned short hb_lds[16][DC_];
  __shared__ float g_lds[16][G4C];

  const int tid = threadIdx.x;
  const int lane = tid & 31;
  const int wv = tid >> 5;           // 8 waves
  const int s0 = blockIdx.x * 16;    // first word of this tile

  for (int i = tid; i < CV_ * DC_; i += 256) ce_lds[i] = f2bf(char_emb[i]);
  for (int i = tid; i < 16 * L_; i += 256)
    cid_lds[i >> 4][i & 15] = char_ids[(size_t)(s0 + (i >> 4)) * L_ + (i & 15)];
  if (tid < 16) wlen_lds[tid] = word_lengths[s0 + tid];
  for (int i = tid; i < 16 * DC_; i += 256) {
    const int m = i >> 6, ch = i & 63;
    h_lds[m][ch] = 0.f; c_lds[m][ch] = 0.f; hb_lds[m][ch] = 0;
  }
  __syncthreads();

  const int nt0 = wv * 2, nt1 = wv * 2 + 1;
  const int col0 = nt0 * 16 + (lane & 15);
  const int col1 = nt1 * 16 + (lane & 15);
  FragU bih0[2], bih1[2], bhh0[2], bhh1[2];
#pragma unroll
  for (int kc = 0; kc < 2; ++kc) {
    bih0[kc] = load_bfrag_f32(cWih, DC_, col0, kc, lane);
    bih1[kc] = load_bfrag_f32(cWih, DC_, col1, kc, lane);
    bhh0[kc] = load_bfrag_f32(cWhh, DC_, col0, kc, lane);
    bhh1[kc] = load_bfrag_f32(cWhh, DC_, col1, kc, lane);
  }
  const float bias0 = cb[col0], bias1 = cb[col1];
  const int arow = lane & 15;
  const int rowoff = (lane & 16) ? 8 : 0;

  for (int t = 0; t < L_; ++t) {
    const int cid = cid_lds[arow][t];
    FragU ax0 = load_afrag_u16(ce_lds, DC_, cid, 0, lane);
    FragU ax1 = load_afrag_u16(ce_lds, DC_, cid, 1, lane);
    FragU ah0 = load_afrag_u16(&hb_lds[0][0], DC_, arow, 0, lane);
    FragU ah1 = load_afrag_u16(&hb_lds[0][0], DC_, arow, 1, lane);

    v8f acc0, acc1;
#pragma unroll
    for (int e = 0; e < 8; ++e) { acc0[e] = bias0; acc1[e] = bias1; }
    acc0 = WMMA_BF16(ax0.v, bih0[0].v, acc0);
    acc0 = WMMA_BF16(ax1.v, bih0[1].v, acc0);
    acc0 = WMMA_BF16(ah0.v, bhh0[0].v, acc0);
    acc0 = WMMA_BF16(ah1.v, bhh0[1].v, acc0);
    acc1 = WMMA_BF16(ax0.v, bih1[0].v, acc1);
    acc1 = WMMA_BF16(ax1.v, bih1[1].v, acc1);
    acc1 = WMMA_BF16(ah0.v, bhh1[0].v, acc1);
    acc1 = WMMA_BF16(ah1.v, bhh1[1].v, acc1);
#pragma unroll
    for (int r = 0; r < 8; ++r) {
      g_lds[r + rowoff][col0] = acc0[r];
      g_lds[r + rowoff][col1] = acc1[r];
    }
    __syncthreads();
#pragma unroll
    for (int k = 0; k < 4; ++k) {
      const int idx = tid + 256 * k;
      const int m = idx >> 6, ch = idx & 63;
      const float gi = g_lds[m][ch];
      const float gf = g_lds[m][DC_ + ch];
      const float gg = g_lds[m][2 * DC_ + ch];
      const float go = g_lds[m][3 * DC_ + ch];
      const float cold = c_lds[m][ch], hold = h_lds[m][ch];
      float cn = sigm(gf) * cold + sigm(gi) * fast_tanh(gg);
      float hn = sigm(go) * fast_tanh(cn);
      const bool valid = t < wlen_lds[m];
      cn = valid ? cn : cold;
      hn = valid ? hn : hold;
      c_lds[m][ch] = cn; h_lds[m][ch] = hn; hb_lds[m][ch] = f2bf(hn);
    }
    __syncthreads();
  }
  for (int i = tid; i < 16 * DC_; i += 256) {
    const int m = i >> 6, ch = i & 63;
    wx[(size_t)(s0 + m) * WXLD + DW_ + ch] = f2bf(h_lds[m][ch]);
  }
}

// ========= kernel 3: x-gates precompute  Xg = wx @ wWih^T + wb  (WMMA GEMM) =========
__global__ __launch_bounds__(256) void xgates_kernel(
    const unsigned short* __restrict__ wx, const float* __restrict__ wWih,
    const float* __restrict__ wb, float* __restrict__ xg) {
  const int tid = threadIdx.x, lane = tid & 31, wv = tid >> 5;
  const int mb = blockIdx.x, ng = blockIdx.y;
  const int nt = ng * 8 + wv;
  const int col = nt * 16 + (lane & 15);
  const int row = mb * 16 + (lane & 15);

  v8f acc;
  const float bias = wb[col];
#pragma unroll
  for (int e = 0; e < 8; ++e) acc[e] = bias;
#pragma unroll
  for (int kc = 0; kc < 6; ++kc) {
    FragU a = load_afrag_u16(wx, WXLD, row, kc, lane);
    FragU b = load_bfrag_f32(wWih, WXLD, col, kc, lane);
    acc = WMMA_BF16(a.v, b.v, acc);
  }
  const int rowoff = (lane & 16) ? 8 : 0;
#pragma unroll
  for (int r = 0; r < 8; ++r)
    xg[(size_t)(mb * 16 + r + rowoff) * G4H + col] = acc[r];
}

// ========= kernel 4: word recurrence, single persistent workgroup =========
// wWhh (bf16) lives in VGPRs as prebuilt B fragments; per step: 16 WMMAs (row 0
// live), gate fuse with precomputed x-gates, h/c update. The next step's 4KB
// x-gate row is DMA'd into an LDS double buffer by the TDM (tensor_load_to_lds,
// TENSORcnt-tracked), overlapping with this step's WMMAs.
__global__ __launch_bounds__(1024, 1) void word_rnn_kernel(
    const float* __restrict__ xg, const float* __restrict__ wWhh,
    float* __restrict__ hs) {
  __shared__ __align__(16) unsigned short hb[H_];
  __shared__ float g_all[G4H];
  __shared__ __align__(16) float xg_lds[2][G4H];   // TDM double buffer

  const int tid = threadIdx.x, lane = tid & 31, wv = tid >> 5;  // 32 waves
  const int col0 = wv * 32 + (lane & 15);
  const int col1 = wv * 32 + 16 + (lane & 15);

  FragU B0[8], B1[8];
#pragma unroll
  for (int kc = 0; kc < 8; ++kc) {
    B0[kc] = load_bfrag_f32(wWhh, H_, col0, kc, lane);
    B1[kc] = load_bfrag_f32(wWhh, H_, col1, kc, lane);
  }
  float c_reg = 0.f;
  if (tid < H_) hb[tid] = 0;

  // --- TDM descriptor pieces (wave-uniform). Generic LDS pointers carry the
  // wave-relative LDS byte offset in their low 32 bits. ---
  const unsigned lds_buf0 = (unsigned)(uintptr_t)&xg_lds[0][0];
  const unsigned lds_buf1 = (unsigned)(uintptr_t)&xg_lds[1][0];
  const unsigned long long gbase = (unsigned long long)(uintptr_t)xg;

  // Group1: workgroup_mask=0, data_size=2 (4B), no barrier/iterate/pad,
  // tensor_dim0 = tile_dim0 = 1024 elements (one x-gate row), 1-D tile.
  i32x8 g1;
  g1[0] = 0x00020000;           // data_size = 4B
  g1[1] = (int)(1024u << 16);   // tensor_dim0[15:0] in bits 63:48
  g1[2] = 0;                    // tensor_dim0 hi / tensor_dim1 lo
  g1[3] = (int)(1024u << 16);   // tile_dim0 in bits 127:112
  g1[4] = 0;                    // tile_dim1 = 0 (unused), tile_dim2 = 0
  g1[5] = 1024;                 // tensor_dim0_stride lo
  g1[6] = 0; g1[7] = 0;
  i32x4 gz; gz[0] = 0; gz[1] = 0; gz[2] = 0; gz[3] = 0;
  i32x8 gz8;
#pragma unroll
  for (int e = 0; e < 8; ++e) gz8[e] = 0;

  if (wv == 0) {
    // prologue: DMA step-0 row into buffer 0, wait for it
    u32x4 g0;
    g0[0] = 1u;                                   // count=1, user mode
    g0[1] = lds_buf0;
    g0[2] = (unsigned)(gbase & 0xFFFFFFFFu);
    g0[3] = (unsigned)((gbase >> 32) & 0x1FFFFFFu) | (2u << 30);  // type=2
    __builtin_amdgcn_tensor_load_to_lds(g0, g1, gz, gz, gz8, 0);
    __builtin_amdgcn_s_wait_tensorcnt(0);
  }
  __syncthreads();

  const bool rowlane = ((lane & 15) == 0);       // lanes 0 and 16 carry row 0
  const int kboff = (lane & 16) ? 8 : 0;

  for (int s = 0; s < S_; ++s) {
    const int p = s & 1;
    if (wv == 0 && s + 1 < S_) {
      const unsigned long long ga = gbase + (unsigned long long)(s + 1) * (G4H * 4);
      u32x4 g0;
      g0[0] = 1u;
      g0[1] = p ? lds_buf0 : lds_buf1;
      g0[2] = (unsigned)(ga & 0xFFFFFFFFu);
      g0[3] = (unsigned)((ga >> 32) & 0x1FFFFFFu) | (2u << 30);
      __builtin_amdgcn_tensor_load_to_lds(g0, g1, gz, gz, gz8, 0);
    }

    v8f acc0, acc1;
#pragma unroll
    for (int e = 0; e < 8; ++e) { acc0[e] = 0.f; acc1[e] = 0.f; }
#pragma unroll
    for (int kc = 0; kc < 8; ++kc) {
      FragU a;
      if (rowlane) {
        const unsigned short* q = &hb[kc * 32 + kboff];
        a.q[0] = *(const uint4*)(q);
        a.q[1] = *(const uint4*)(q + 16);
      } else {
        a.q[0] = make_uint4(0u, 0u, 0u, 0u);
        a.q[1] = make_uint4(0u, 0u, 0u, 0u);
      }
      acc0 = WMMA_BF16(a.v, B0[kc].v, acc0);
      acc1 = WMMA_BF16(a.v, B1[kc].v, acc1);
    }
    if (lane < 16) {
      g_all[col0] = acc0[0] + xg_lds[p][col0];
      g_all[col1] = acc1[0] + xg_lds[p][col1];
    }
    __syncthreads();
    if (tid < H_) {
      const float gi = g_all[tid];
      const float gf = g_all[H_ + tid];
      const float gg = g_all[2 * H_ + tid];
      const float go = g_all[3 * H_ + tid];
      c_reg = sigm(gf) * c_reg + sigm(gi) * fast_tanh(gg);
      const float h = sigm(go) * fast_tanh(c_reg);
      hs[(size_t)s * H_ + tid] = h;
      hb[tid] = f2bf(h);
    }
    if (wv == 0) __builtin_amdgcn_s_wait_tensorcnt(0);
    __syncthreads();
  }
}

// ========= kernel 5: output projection + log_softmax (wave per row) =========
__global__ __launch_bounds__(256) void proj_kernel(
    const float* __restrict__ hs, const float* __restrict__ outW,
    const float* __restrict__ outb, float* __restrict__ out) {
  __shared__ float rowb[8][H_];
  const int tid = threadIdx.x, lane = tid & 31, wv = tid >> 5;
  const int s = blockIdx.x * 8 + wv;
#pragma unroll
  for (int i = 0; i < 8; ++i)
    rowb[wv][lane * 8 + i] = hs[(size_t)s * H_ + lane * 8 + i];
  __syncthreads();

  const int c0 = lane, c1 = lane + 32;
  float x0 = -1e30f, x1 = -1e30f;
  if (c0 < T_) {
    float d = outb[c0];
    for (int k = 0; k < H_; ++k) d += rowb[wv][k] * outW[(size_t)c0 * H_ + k];
    x0 = d;
  }
  if (c1 < T_) {
    float d = outb[c1];
    for (int k = 0; k < H_; ++k) d += rowb[wv][k] * outW[(size_t)c1 * H_ + k];
    x1 = d;
  }
  float m = fmaxf(x0, x1);
  for (int off = 16; off; off >>= 1) m = fmaxf(m, __shfl_xor(m, off, 32));
  float e = (c0 < T_ ? __expf(x0 - m) : 0.f) + (c1 < T_ ? __expf(x1 - m) : 0.f);
  for (int off = 16; off; off >>= 1) e += __shfl_xor(e, off, 32);
  const float lse = m + __logf(e);
  if (c0 < T_) out[(size_t)s * T_ + c0] = x0 - lse;
  if (c1 < T_) out[(size_t)s * T_ + c1] = x1 - lse;
}

// ================= host-side launch =================
extern "C" void kernel_launch(void* const* d_in, const int* in_sizes, int n_in,
                              void* d_out, int out_size, void* d_ws, size_t ws_size,
                              hipStream_t stream) {
  (void)in_sizes; (void)n_in; (void)out_size; (void)ws_size;
  const int*   word_ids     = (const int*)d_in[0];
  const int*   char_ids     = (const int*)d_in[1];
  const int*   word_lengths = (const int*)d_in[2];
  const float* word_emb     = (const float*)d_in[3];
  const float* char_emb     = (const float*)d_in[4];
  const float* cWih         = (const float*)d_in[5];
  const float* cWhh         = (const float*)d_in[6];
  const float* cbih         = (const float*)d_in[7];
  const float* cbhh         = (const float*)d_in[8];
  const float* wWih         = (const float*)d_in[9];
  const float* wWhh         = (const float*)d_in[10];
  const float* wbih         = (const float*)d_in[11];
  const float* wbhh         = (const float*)d_in[12];
  const float* outW         = (const float*)d_in[13];
  const float* outb         = (const float*)d_in[14];

  char* ws = (char*)d_ws;
  unsigned short* wx = (unsigned short*)(ws + WS_WX);
  float* xg = (float*)(ws + WS_XG);
  float* hs = (float*)(ws + WS_HS);
  float* cb = (float*)(ws + WS_CB);
  float* wb = (float*)(ws + WS_WB);
  float* out = (float*)d_out;

  prep_kernel<<<1024, 256, 0, stream>>>(word_ids, word_emb, cbih, cbhh, wbih, wbhh,
                                        wx, cb, wb);
  char_lstm_kernel<<<S_ / 16, 256, 0, stream>>>(char_ids, word_lengths, char_emb,
                                                cWih, cWhh, cb, wx);
  xgates_kernel<<<dim3(S_ / 16, 8), 256, 0, stream>>>(wx, wWih, wb, xg);
  word_rnn_kernel<<<1, 1024, 0, stream>>>(xg, wWhh, hs);
  proj_kernel<<<S_ / 8, 256, 0, stream>>>(hs, outW, outb, out);
}
